// SCNLayer_9466107920681
// MI455X (gfx1250) — matrix operand order repaired and checked by reading, not compiled, not run on hardware
//
#include <hip/hip_runtime.h>

typedef __attribute__((ext_vector_type(16))) _Float16 v16h;
typedef __attribute__((ext_vector_type(8)))  _Float16 v8h;
typedef __attribute__((ext_vector_type(8)))  float    v8f;
typedef __attribute__((ext_vector_type(4)))  float    v4f;

#define N0 8192
#define N1 12288
#define N2 4096
#define E0 (N0 * 128)
#define E1 (N1 * 128)
#define E2 (N2 * 128)
#define EW (128 * 128)

// ---------------------------------------------------------------------------
// Single fused prep kernel:
//   x_i (N x 128 f32) -> xT_i (128 x N f16, transposed: B-fragment friendly)
//   W_i (128x128 f32) -> Wh_i (f16, same layout)
// Flat index space over all six conversions -> one launch.
// ---------------------------------------------------------------------------
__global__ void prep_all(const float* __restrict__ x0, const float* __restrict__ x1,
                         const float* __restrict__ x2, const float* __restrict__ W0,
                         const float* __restrict__ W1, const float* __restrict__ W2,
                         _Float16* __restrict__ xT0, _Float16* __restrict__ xT1,
                         _Float16* __restrict__ xT2, _Float16* __restrict__ Wh0,
                         _Float16* __restrict__ Wh1, _Float16* __restrict__ Wh2) {
  int idx = blockIdx.x * blockDim.x + threadIdx.x;
  if (idx < E0) {
    int n = idx >> 7, c = idx & 127;
    xT0[(size_t)c * N0 + n] = (_Float16)x0[idx];
    return;
  }
  idx -= E0;
  if (idx < E1) {
    int n = idx >> 7, c = idx & 127;
    xT1[(size_t)c * N1 + n] = (_Float16)x1[idx];
    return;
  }
  idx -= E1;
  if (idx < E2) {
    int n = idx >> 7, c = idx & 127;
    xT2[(size_t)c * N2 + n] = (_Float16)x2[idx];
    return;
  }
  idx -= E2;
  if (idx < EW) { Wh0[idx] = (_Float16)W0[idx]; return; }
  idx -= EW;
  if (idx < EW) { Wh1[idx] = (_Float16)W1[idx]; return; }
  idx -= EW;
  if (idx < EW) { Wh2[idx] = (_Float16)W2[idx]; return; }
}

// ---------------------------------------------------------------------------
// One branch body: y = relu((L @ x + x) @ W^T + b)
// Each wave computes 16 rows x 128 cols. Block = 8 waves = 128 rows.
// ---------------------------------------------------------------------------
__device__ __forceinline__ void scn_branch_body(
    const float* __restrict__ L, const float* __restrict__ x,
    const _Float16* __restrict__ xT, const _Float16* __restrict__ Wh,
    const float* __restrict__ bias, float* __restrict__ y, int N, int blk,
    _Float16* lds) {
  const int lane = threadIdx.x & 31;
  const int wave = threadIdx.x >> 5;
  const int half = lane >> 4;  // 0: lanes 0-15, 1: lanes 16-31
  const int l16  = lane & 15;
  const int row0 = (blk * 8 + wave) * 16;

  v8f acc[8] = {};  // 8 col-tiles of 16x16 f32 accumulators

  // A-fragment (16x32 f16) layout: lane l (row = l%16) holds
  // K = kb + half*16 + {0..15}.
  const float* Arow = L + (size_t)(row0 + l16) * N + half * 16;

  for (int kb = 0; kb < N; kb += 32) {
    // Each lane prefetches exactly the 128B line its row consumes 32 k-steps
    // later (speculative; dropped harmlessly past the end of L).
    __builtin_prefetch(Arow + kb + 1024);

    v4f a0 = __builtin_nontemporal_load((const v4f*)(Arow + kb));
    v4f a1 = __builtin_nontemporal_load((const v4f*)(Arow + kb + 4));
    v4f a2 = __builtin_nontemporal_load((const v4f*)(Arow + kb + 8));
    v4f a3 = __builtin_nontemporal_load((const v4f*)(Arow + kb + 12));
    v16h af;
#pragma unroll
    for (int i = 0; i < 4; ++i) {
      af[i]      = (_Float16)a0[i];
      af[4 + i]  = (_Float16)a1[i];
      af[8 + i]  = (_Float16)a2[i];
      af[12 + i] = (_Float16)a3[i];
    }
#pragma unroll
    for (int ct = 0; ct < 8; ++ct) {
      // B-fragment (32x16 f16): lane n = l%16 holds column ct*16+n,
      // K = kb + half*16 + {0..15}: one contiguous 32-byte read of xT.
      v16h bf = *(const v16h*)(xT + (size_t)(ct * 16 + l16) * N + kb + half * 16);
      acc[ct] = __builtin_amdgcn_wmma_f32_16x16x32_f16(
          false, af, false, bf, (short)0, acc[ct], false, false);
    }
  }

  // Epilogue part 1: prop = acc + x, dumped to LDS as f16.
  // C-fragment: VGPR r, lane -> M = r + 8*half, Ncol = ct*16 + l16.
  _Float16* myTile = lds + wave * (16 * 136);
#pragma unroll
  for (int ct = 0; ct < 8; ++ct) {
#pragma unroll
    for (int r = 0; r < 8; ++r) {
      int m = r + 8 * half;
      int n = ct * 16 + l16;
      float v = acc[ct][r] + x[(size_t)(row0 + m) * 128 + n];
      myTile[m * 136 + n] = (_Float16)v;
    }
  }
  __syncthreads();

  // Epilogue part 2: y = relu(prop @ W^T + b), K = 128 in 4 steps of 32.
  v8f out[8] = {};
  const _Float16* arow = myTile + l16 * 136 + half * 16;
#pragma unroll
  for (int kc = 0; kc < 128; kc += 32) {
    v8h p0 = *(const v8h*)(arow + kc);
    v8h p1 = *(const v8h*)(arow + kc + 8);
    v16h af;
#pragma unroll
    for (int i = 0; i < 8; ++i) { af[i] = p0[i]; af[8 + i] = p1[i]; }
#pragma unroll
    for (int ot = 0; ot < 8; ++ot) {
      v16h bf = *(const v16h*)(Wh + (size_t)(ot * 16 + l16) * 128 + kc + half * 16);
      out[ot] = __builtin_amdgcn_wmma_f32_16x16x32_f16(
          false, af, false, bf, (short)0, out[ot], false, false);
    }
  }

#pragma unroll
  for (int ot = 0; ot < 8; ++ot) {
    float bb = bias[ot * 16 + l16];
#pragma unroll
    for (int r = 0; r < 8; ++r) {
      float v = out[ot][r] + bb;
      v = v > 0.0f ? v : 0.0f;
      y[(size_t)(row0 + r + 8 * half) * 128 + ot * 16 + l16] = v;
    }
  }
}

// ---------------------------------------------------------------------------
// Fused kernel: all three branches in one launch so they run concurrently and
// fill the chip. Blocks [0,64) -> branch0, [64,160) -> branch1, [160,192) ->
// branch2. Branch selection is uniform per block.
// ---------------------------------------------------------------------------
__global__ __launch_bounds__(256) void scn_fused(
    const float* __restrict__ L0, const float* __restrict__ x0,
    const _Float16* __restrict__ xT0, const _Float16* __restrict__ Wh0,
    const float* __restrict__ b0, float* __restrict__ y0,
    const float* __restrict__ L1, const float* __restrict__ x1,
    const _Float16* __restrict__ xT1, const _Float16* __restrict__ Wh1,
    const float* __restrict__ b1, float* __restrict__ y1,
    const float* __restrict__ L2, const float* __restrict__ x2,
    const _Float16* __restrict__ xT2, const _Float16* __restrict__ Wh2,
    const float* __restrict__ b2, float* __restrict__ y2) {
  __shared__ _Float16 lds[8 * 16 * 136];
  int blk = blockIdx.x;
  if (blk < N0 / 128) {
    scn_branch_body(L0, x0, xT0, Wh0, b0, y0, N0, blk, lds);
  } else if (blk < N0 / 128 + N1 / 128) {
    scn_branch_body(L1, x1, xT1, Wh1, b1, y1, N1, blk - N0 / 128, lds);
  } else {
    scn_branch_body(L2, x2, xT2, Wh2, b2, y2, N2,
                    blk - N0 / 128 - N1 / 128, lds);
  }
}

// ---------------------------------------------------------------------------
extern "C" void kernel_launch(void* const* d_in, const int* in_sizes, int n_in,
                              void* d_out, int out_size, void* d_ws,
                              size_t ws_size, hipStream_t stream) {
  const float* x0 = (const float*)d_in[0];
  const float* x1 = (const float*)d_in[1];
  const float* x2 = (const float*)d_in[2];
  const float* L0 = (const float*)d_in[3];
  const float* L1 = (const float*)d_in[4];
  const float* L2 = (const float*)d_in[5];
  const float* W0 = (const float*)d_in[6];
  const float* b0 = (const float*)d_in[7];
  const float* W1 = (const float*)d_in[8];
  const float* b1 = (const float*)d_in[9];
  const float* W2 = (const float*)d_in[10];
  const float* b2 = (const float*)d_in[11];
  float* y = (float*)d_out;

  char* ws = (char*)d_ws;
  _Float16* xT0 = (_Float16*)ws; ws += (size_t)E0 * sizeof(_Float16);
  _Float16* xT1 = (_Float16*)ws; ws += (size_t)E1 * sizeof(_Float16);
  _Float16* xT2 = (_Float16*)ws; ws += (size_t)E2 * sizeof(_Float16);
  _Float16* Wh0 = (_Float16*)ws; ws += EW * sizeof(_Float16);
  _Float16* Wh1 = (_Float16*)ws; ws += EW * sizeof(_Float16);
  _Float16* Wh2 = (_Float16*)ws; ws += EW * sizeof(_Float16);

  const int prepElems = E0 + E1 + E2 + 3 * EW;
  prep_all<<<(prepElems + 255) / 256, 256, 0, stream>>>(
      x0, x1, x2, W0, W1, W2, xT0, xT1, xT2, Wh0, Wh1, Wh2);

  const int nBlocks = N0 / 128 + N1 / 128 + N2 / 128;  // 64 + 96 + 32 = 192
  scn_fused<<<nBlocks, 256, 0, stream>>>(L0, x0, xT0, Wh0, b0, y,
                                         L1, x1, xT1, Wh1, b1, y + (size_t)E0,
                                         L2, x2, xT2, Wh2, b2,
                                         y + (size_t)(E0 + E1));
}
